// CollisonToJointLoss_79070347920142
// MI455X (gfx1250) — matrix operand order repaired and checked by reading, not compiled, not run on hardware
//
#include <hip/hip_runtime.h>
#include <math.h>

// Problem constants (match reference)
#define Bb 16
#define Cc 2048
#define Nn 6890
#define Ff 13776
#define Jj 24

#define K3               (Bb * Cc * 3)          // 98304 collision-vertices
#define STEPS            (K3 / 32)              // 3072 k-steps of 32
#define STEPS_PER_BATCH  (STEPS / Bb)           // 192 (batch-aligned: 6144/32)
#define WAVES_PER_BATCH  48
#define STEPS_PER_WAVE   (STEPS_PER_BATCH / WAVES_PER_BATCH)  // 4
#define WPB              8                       // waves per 256-thread block
#define BLOCKS_PER_BATCH (WAVES_PER_BATCH / WPB) // 6
#define NWAVES           (Bb * WAVES_PER_BATCH)  // 768 partials

typedef __attribute__((ext_vector_type(16))) _Float16 v16h;
typedef __attribute__((ext_vector_type(8)))  float    v8f;

// ---------------------------------------------------------------------------
// Kernel 1: transpose joint_regressor [J,N] (f32) -> jrT [N,32] (f16, padded 0)
// ---------------------------------------------------------------------------
__global__ void jrt_kernel(const float* __restrict__ jr, _Float16* __restrict__ jrT) {
  int tid = blockIdx.x * blockDim.x + threadIdx.x;
  if (tid >= Nn * 32) return;
  int n = tid >> 5, j = tid & 31;
  float v = (j < Jj) ? jr[j * Nn + n] : 0.0f;
  jrT[tid] = (_Float16)v;
}

// ---------------------------------------------------------------------------
// Kernel 2: joints[b,j,:] = sum_n vertices[b,n,:] * jr[j,n]   (one wave each)
// ---------------------------------------------------------------------------
__global__ void joints_kernel(const float* __restrict__ vertices,
                              const float* __restrict__ jr,
                              float* __restrict__ joints) {
  int wave = blockIdx.x * (blockDim.x >> 5) + (threadIdx.x >> 5);
  int lane = threadIdx.x & 31;
  if (wave >= Bb * Jj) return;
  int b = wave / Jj, j = wave % Jj;
  float a0 = 0.f, a1 = 0.f, a2 = 0.f;
  for (int n = lane; n < Nn; n += 32) {
    float w = jr[j * Nn + n];
    if (w != 0.0f) {                       // regressor is ~8% dense
      const float* vp = vertices + ((size_t)b * Nn + n) * 3;
      a0 += w * vp[0]; a1 += w * vp[1]; a2 += w * vp[2];
    }
  }
  for (int off = 16; off; off >>= 1) {
    a0 += __shfl_xor(a0, off, 32);
    a1 += __shfl_xor(a1, off, 32);
    a2 += __shfl_xor(a2, off, 32);
  }
  if (lane == 0) {
    float* jp = joints + (size_t)wave * 3;
    jp[0] = a0; jp[1] = a1; jp[2] = a2;
  }
}

// ---------------------------------------------------------------------------
// Kernel 3: padded pairwise joint distances Dpad[b][32][32]
// ---------------------------------------------------------------------------
__global__ void dists_kernel(const float* __restrict__ joints, float* __restrict__ Dpad) {
  int b = blockIdx.x, tid = threadIdx.x;
  int i = tid >> 5, j = tid & 31;
  float v = 0.f;
  if (i < Jj && j < Jj) {
    const float* ji = joints + ((size_t)b * Jj + i) * 3;
    const float* jp = joints + ((size_t)b * Jj + j) * 3;
    float dx = jp[0] - ji[0], dy = jp[1] - ji[1], dz = jp[2] - ji[2];
    v = sqrtf(dx * dx + dy * dy + dz * dz);
  }
  Dpad[(size_t)b * 1024 + tid] = v;
}

// ---------------------------------------------------------------------------
// Kernel 4: producer-side swizzle into WMMA-native f16 fragments.
//   Afrag = U (intr scores, J x m) in 16-bit A 16x32 layout (ISA 7.12.2)
//   Pfrag = indicator(U != 0) in the same A layout
//   Bfrag = V (recv scores, m x J) in 16-bit B 32x16 layout
//   Qfrag = indicator(V != 0) in the same B layout
// Invalid collisions (cf0 < 0) -> score rows zeroed, so they vanish from
// both num and den without any masking downstream.
// ---------------------------------------------------------------------------
__device__ __forceinline__ _Float16 score_at(const int* __restrict__ ci,
                                             const int* __restrict__ faces,
                                             const _Float16* __restrict__ jrT,
                                             int m, int j, bool intr_side) {
  int k = m / 3, t = m - 3 * k;      // collision index, vertex-in-face
  int b = k / Cc;
  int cf0 = ci[2 * k];
  if (cf0 < 0) return (_Float16)0.0f;  // invalid collision
  int f = intr_side ? ci[2 * k + 1] : cf0;
  if (f < 0) f = 0;                    // clip (matches reference)
  int v = faces[(b * Ff + f) * 3 + t];
  return jrT[v * 32 + j];
}

__global__ void pack_kernel(const int* __restrict__ ci, const int* __restrict__ faces,
                            const _Float16* __restrict__ jrT,
                            v16h* __restrict__ Afrag, v16h* __restrict__ Pfrag,
                            v16h* __restrict__ Bfrag, v16h* __restrict__ Qfrag) {
  int tid = blockIdx.x * blockDim.x + threadIdx.x;
  int step = tid >> 5, lane = tid & 31;
  if (step >= STEPS) return;
  int m_base = step * 32;
  int jl = lane & 15;
  bool hi = lane >= 16;

  #pragma unroll
  for (int t = 0; t < 2; ++t) {          // J tile: rows 0-15 / 16-31
    int j = t * 16 + jl;
    v16h a, p, v, q;
    #pragma unroll
    for (int w = 0; w < 8; ++w) {
      // A 16x32 f16: lanes 0-15 hold K {0..7,16..23}, lanes 16-31 hold +8
      int kbA = (w < 4 ? 2 * w : 16 + 2 * (w - 4)) + (hi ? 8 : 0);
      // B 32x16 f16: same K order as A (lanes 0-15 hold K {0..7,16..23}, lanes 16-31 hold +8)
      int kbB = (w < 4 ? 2 * w : 16 + 2 * (w - 4)) + (hi ? 8 : 0);
      #pragma unroll
      for (int e = 0; e < 2; ++e) {
        _Float16 av = score_at(ci, faces, jrT, m_base + kbA + e, j, true);
        _Float16 bv = score_at(ci, faces, jrT, m_base + kbB + e, j, false);
        a[2 * w + e] = av;
        p[2 * w + e] = (av != (_Float16)0.0f) ? (_Float16)1.0f : (_Float16)0.0f;
        v[2 * w + e] = bv;
        q[2 * w + e] = (bv != (_Float16)0.0f) ? (_Float16)1.0f : (_Float16)0.0f;
      }
    }
    int idx = (step * 2 + t) * 32 + lane;
    Afrag[idx] = a;
    Pfrag[idx] = p;
    Bfrag[idx] = v;
    Qfrag[idx] = q;
  }
}

// ---------------------------------------------------------------------------
// Kernel 5: the WMMA workhorse.  Per wave, over its k-slice:
//   Cn += U*Q^T + P*V^T   (num GEMM),   Cd += P*Q^T   (den GEMM)
// unroll(1): one step's fragments (64 VGPRs) + 8 accumulators (64 VGPRs)
// stay register-resident -> no scratch spills; step+1 is prefetched at
// WGP scope (locality 3) so loads hit near caches next iteration.
// ---------------------------------------------------------------------------
__global__ void wmma_kernel(const v16h* __restrict__ Afrag, const v16h* __restrict__ Pfrag,
                            const v16h* __restrict__ Bfrag, const v16h* __restrict__ Qfrag,
                            const float* __restrict__ Dpad, float* __restrict__ part) {
  int wave = threadIdx.x >> 5, lane = threadIdx.x & 31;
  int b = blockIdx.x / BLOCKS_PER_BATCH;
  int wslot = (blockIdx.x % BLOCKS_PER_BATCH) * WPB + wave;      // 0..47
  int step0 = b * STEPS_PER_BATCH + wslot * STEPS_PER_WAVE;

  v8f Cn[2][2] = {};
  v8f Cd[2][2] = {};

  #pragma unroll 1
  for (int s = 0; s < STEPS_PER_WAVE; ++s) {
    int step = step0 + s;
    {   // speculative prefetch of next step (dropped harmlessly past the end)
      int nidx = ((step + 1) * 2) * 32 + lane;
      __builtin_prefetch((const void*)&Afrag[nidx], 0, 3);
      __builtin_prefetch((const void*)&Pfrag[nidx], 0, 3);
      __builtin_prefetch((const void*)&Bfrag[nidx], 0, 3);
      __builtin_prefetch((const void*)&Qfrag[nidx], 0, 3);
    }
    v16h A[2], P[2], Bv[2], Q[2];
    #pragma unroll
    for (int t = 0; t < 2; ++t) {
      int idx = (step * 2 + t) * 32 + lane;
      A[t]  = Afrag[idx];
      P[t]  = Pfrag[idx];
      Bv[t] = Bfrag[idx];
      Q[t]  = Qfrag[idx];
    }
    #pragma unroll
    for (int ti = 0; ti < 2; ++ti)
      #pragma unroll
      for (int tj = 0; tj < 2; ++tj) {
        Cn[ti][tj] = __builtin_amdgcn_wmma_f32_16x16x32_f16(
            false, A[ti], false, Q[tj], (short)0, Cn[ti][tj], false, false);
        Cn[ti][tj] = __builtin_amdgcn_wmma_f32_16x16x32_f16(
            false, P[ti], false, Bv[tj], (short)0, Cn[ti][tj], false, false);
        Cd[ti][tj] = __builtin_amdgcn_wmma_f32_16x16x32_f16(
            false, P[ti], false, Q[tj], (short)0, Cd[ti][tj], false, false);
      }
  }

  // Frobenius contraction <D_b, Cn> and sum(Cd).
  // f32 C/D layout: VGPR r, lane l: M = (l<16 ? r : r+8), N = l&15.
  const float* Db = Dpad + (size_t)b * 1024;
  int jloc  = lane & 15;
  int ibase = (lane < 16) ? 0 : 8;
  float num = 0.f, den = 0.f;
  #pragma unroll
  for (int ti = 0; ti < 2; ++ti)
    #pragma unroll
    for (int tj = 0; tj < 2; ++tj)
      #pragma unroll
      for (int r = 0; r < 8; ++r) {
        int i = ti * 16 + ibase + r;
        int j = tj * 16 + jloc;
        num += Cn[ti][tj][r] * Db[i * 32 + j];
        den += Cd[ti][tj][r];
      }
  for (int off = 16; off; off >>= 1) {
    num += __shfl_xor(num, off, 32);
    den += __shfl_xor(den, off, 32);
  }
  if (lane == 0) {
    int w = blockIdx.x * WPB + wave;      // deterministic slot, no atomics
    part[2 * w + 0] = num;
    part[2 * w + 1] = den;
  }
}

// ---------------------------------------------------------------------------
// Kernel 6: deterministic fixed-order reduction of per-wave partials.
// ---------------------------------------------------------------------------
__global__ void finalize_kernel(const float* __restrict__ part, float* __restrict__ out) {
  float num = 0.f, den = 0.f;
  for (int i = 0; i < NWAVES; ++i) { num += part[2 * i]; den += part[2 * i + 1]; }
  out[0] = (den > 0.f) ? (num / fmaxf(den, 1.0f)) : 0.0f;
}

// ---------------------------------------------------------------------------
extern "C" void kernel_launch(void* const* d_in, const int* in_sizes, int n_in,
                              void* d_out, int out_size, void* d_ws, size_t ws_size,
                              hipStream_t stream) {
  (void)in_sizes; (void)n_in; (void)out_size; (void)ws_size;
  const int*   collision = (const int*)d_in[0];    // [B,C,2]
  const float* vertices  = (const float*)d_in[1];  // [B,N,3]
  const int*   faces     = (const int*)d_in[2];    // [B,F,3]
  const float* jr        = (const float*)d_in[3];  // [J,N]
  float* out = (float*)d_out;

  // Workspace carving (256B aligned); total ~25.7 MB, fully L2-resident.
  char* p = (char*)d_ws;
  auto take = [&](size_t bytes) -> void* {
    void* r = (void*)p;
    p += (bytes + 255) & ~(size_t)255;
    return r;
  };
  const size_t frag_bytes = (size_t)STEPS * 2 * 32 * sizeof(v16h);
  _Float16* jrT    = (_Float16*)take((size_t)Nn * 32 * sizeof(_Float16));
  float*    joints = (float*)   take((size_t)Bb * Jj * 3 * sizeof(float));
  float*    Dpad   = (float*)   take((size_t)Bb * 1024 * sizeof(float));
  v16h*     Afrag  = (v16h*)    take(frag_bytes);
  v16h*     Pfrag  = (v16h*)    take(frag_bytes);
  v16h*     Bfrag  = (v16h*)    take(frag_bytes);
  v16h*     Qfrag  = (v16h*)    take(frag_bytes);
  float*    part   = (float*)   take((size_t)NWAVES * 2 * sizeof(float));

  jrt_kernel   <<<(Nn * 32 + 255) / 256, 256, 0, stream>>>(jr, jrT);
  joints_kernel<<<(Bb * Jj) / WPB,       256, 0, stream>>>(vertices, jr, joints);
  dists_kernel <<<Bb,                   1024, 0, stream>>>(joints, Dpad);
  pack_kernel  <<<(STEPS * 32) / 256,    256, 0, stream>>>(collision, faces, jrT,
                                                           Afrag, Pfrag, Bfrag, Qfrag);
  wmma_kernel  <<<Bb * BLOCKS_PER_BATCH, 256, 0, stream>>>(Afrag, Pfrag, Bfrag, Qfrag,
                                                           Dpad, part);
  finalize_kernel<<<1, 1, 0, stream>>>(part, out);
}
